// GroupedQueryAttention_18657337933979
// MI455X (gfx1250) — matrix operand order repaired and checked by reading, not compiled, not run on hardware
//
#include <hip/hip_runtime.h>
#include <hip/hip_bf16.h>
#include <math.h>

// Problem constants (reference: B,T,D = 2,2048,2048; H=32, G=8, DH=64)
#define B_    2
#define T_    2048
#define D_    2048
#define H_    32
#define G_    8
#define DH_   64
#define KVW_  512    // G*DH
#define NQKV_ 3072   // D + 2*KVW
#define TK_   128    // K-slab staged in LDS per GEMM iteration

typedef __bf16 bf16_t;
typedef __attribute__((ext_vector_type(16))) __bf16 v16bf;
typedef __attribute__((ext_vector_type(8)))  __bf16 v8bf;
typedef __attribute__((ext_vector_type(4)))  __bf16 v4bf;
typedef __attribute__((ext_vector_type(8)))  float  v8f;

__device__ __forceinline__ v8f wmma_bf16(v16bf a, v16bf b, v8f c) {
  // 8 args: (neg_a, A, neg_b, B, c_mod, C, reuse_a, reuse_b)
  return __builtin_amdgcn_wmma_f32_16x16x32_bf16(false, a, false, b, (short)0, c,
                                                 false, false);
}

// A-fragment (16-bit 16x32): elems 0..7 at p[0..7], elems 8..15 at p[16..23].
// Caller passes p = rowBase + k0 + kgrpA where kgrpA = (lane>>4)*8.
__device__ __forceinline__ v16bf ld_a_frag(const bf16_t* p) {
  v8bf lo = *(const v8bf*)p;
  v8bf hi = *(const v8bf*)(p + 16);
  v16bf r;
#pragma unroll
  for (int i = 0; i < 8; ++i) { r[i] = lo[i]; r[8 + i] = hi[i]; }
  return r;
}

// B-fragment (16-bit 32x16): lane holds column n; elems 0..7 at p[0..7], elems 8..15 at p[16..23].
// Caller passes p = colBase + k0 + kgrpB where kgrpB = (lane>>4)*8.
__device__ __forceinline__ v16bf ld_b_frag(const bf16_t* p) {
  v8bf lo = *(const v8bf*)p;
  v8bf hi = *(const v8bf*)(p + 16);
  v16bf r;
#pragma unroll
  for (int i = 0; i < 8; ++i) { r[i] = lo[i]; r[8 + i] = hi[i]; }
  return r;
}

// Issue one lane-wide GLOBAL_LOAD_ASYNC_TO_LDS_B128 (gfx1250 async DMA path,
// tracked by ASYNCcnt). ldsOff = 32-bit LDS byte offset, g = global src.
__device__ __forceinline__ void async_copy_b128(unsigned ldsOff, const bf16_t* g) {
  asm volatile("global_load_async_to_lds_b128 %0, %1, off"
               :: "v"(ldsOff), "v"(g) : "memory");
}
__device__ __forceinline__ void wait_asynccnt0() {
  asm volatile("s_wait_asynccnt 0x0" ::: "memory");
}

// ---------------------------------------------------------------------------
// Elementwise converters
// ---------------------------------------------------------------------------
__global__ void cvt2d_f32_bf16(const float* __restrict__ src, bf16_t* __restrict__ dst,
                               int rows, int cols, int srcStride, int srcOff,
                               int dstStride, int dstOff) {
  long long idx = (long long)blockIdx.x * blockDim.x + threadIdx.x;
  long long total = (long long)rows * cols;
  if (idx >= total) return;
  int c = (int)(idx % cols);
  long long r = idx / cols;
  dst[r * (long long)dstStride + dstOff + c] =
      (bf16_t)src[r * (long long)srcStride + srcOff + c];
}

// dst[(rowOff + n) * K + k] = (bf16) src[k * Ncols + n]   (weight transpose)
__global__ void wtrans_f32_bf16(const float* __restrict__ src, bf16_t* __restrict__ dst,
                                int K, int Ncols, int rowOff) {
  long long idx = (long long)blockIdx.x * blockDim.x + threadIdx.x;
  long long total = (long long)K * Ncols;
  if (idx >= total) return;
  int k = (int)(idx % K);
  long long n = idx / K;
  dst[(rowOff + n) * (long long)K + k] = (bf16_t)src[(long long)k * Ncols + n];
}

// V slice of fused QKV -> bf16 transposed layout [b][g][dh][T]
__global__ void vtrans_f32_bf16(const float* __restrict__ qkv, bf16_t* __restrict__ vt) {
  long long idx = (long long)blockIdx.x * blockDim.x + threadIdx.x;
  const long long total = (long long)B_ * G_ * DH_ * T_;
  if (idx >= total) return;
  int t = (int)(idx % T_);
  long long r = idx / T_;
  int dh = (int)(r % DH_); r /= DH_;
  int g  = (int)(r % G_);
  int b  = (int)(r / G_);
  vt[idx] = (bf16_t)qkv[(long long)(b * T_ + t) * NQKV_ + D_ + KVW_ + g * DH_ + dh];
}

__global__ void copy_f32(const float* __restrict__ src, float* __restrict__ dst, int n) {
  int i = blockIdx.x * blockDim.x + threadIdx.x;
  if (i < n) dst[i] = src[i];
}

// ---------------------------------------------------------------------------
// bf16 WMMA GEMM with transposed B:
//   C[MxN] = A[MxK] (row major) * Bt[NxK] (row major, i.e. B column major) + bias
// blockDim = 256 (8 waves). All 8 waves share one 16-row A tile: a 16xTK A slab
// is staged into LDS with GLOBAL_LOAD_ASYNC_TO_LDS_B128 (double-buffered), and
// each wave computes a 16x64 output strip (4 WMMAs per A fragment). B streams
// straight from global as contiguous b128 loads (L2-resident weights).
// Grids must exactly cover M and N (no early-exit: kernel contains barriers).
// ---------------------------------------------------------------------------
__global__ void gemm_bias_bf16t(const bf16_t* __restrict__ A, const bf16_t* __restrict__ Bt,
                                const float* __restrict__ bias, float* __restrict__ C,
                                int M, int N, int K) {
  __shared__ bf16_t shA[2][16 * TK_];   // 2 x 4 KB double buffer

  const int tid   = threadIdx.x;
  const int lane  = tid & 31;
  const int wave  = tid >> 5;
  const int nBase = (blockIdx.x * 8 + wave) * 64;
  const int tm    = blockIdx.y;

  // Cooperative A-slab loader: 256 threads x 16 bytes = 16 rows x 128 cols bf16.
  const int ldRow = tid >> 4;           // 0..15
  const int ldCol = (tid & 15) * 8;     // 0,8,...,120  (bf16 elements)
  const bf16_t* aSrcRow = A + (size_t)(tm * 16 + ldRow) * K + ldCol;
  const unsigned ldsBase0 = (unsigned)(size_t)&shA[0][ldRow * TK_ + ldCol];
  const unsigned ldsBase1 = (unsigned)(size_t)&shA[1][ldRow * TK_ + ldCol];

  const int nloc  = lane & 15;
  const int kgrpA = (lane >> 4) * 8;
  const int kgrpB = (lane >> 4) * 8;
  const bf16_t* bRow = Bt + (size_t)(nBase + nloc) * K;

  // Prologue: stage slab 0.
  async_copy_b128(ldsBase0, aSrcRow);
  wait_asynccnt0();
  __syncthreads();

  v8f acc[4] = {v8f{}, v8f{}, v8f{}, v8f{}};
  for (int k0 = 0; k0 < K; k0 += TK_) {
    const int buf = (k0 / TK_) & 1;
    if (k0 + TK_ < K)   // uniform: prefetch next slab into the other buffer
      async_copy_b128(buf ? ldsBase0 : ldsBase1, aSrcRow + k0 + TK_);

#pragma unroll
    for (int kk = 0; kk < TK_; kk += 32) {
      __builtin_prefetch(bRow + k0 + kk + 512, 0, 3);
      v16bf a = ld_a_frag(&shA[buf][(lane & 15) * TK_ + kk + kgrpA]);
#pragma unroll
      for (int j = 0; j < 4; ++j) {
        v16bf b = ld_b_frag(bRow + (size_t)j * 16 * K + k0 + kk + kgrpB);
        acc[j] = wmma_bf16(a, b, acc[j]);
      }
    }
    wait_asynccnt0();   // next slab resident
    __syncthreads();    // everyone done reading current slab
  }

  const int mbase = tm * 16 + 8 * (lane >> 4);
#pragma unroll
  for (int j = 0; j < 4; ++j) {
    const int ncol = nBase + j * 16 + nloc;
    const float bval = bias ? bias[ncol] : 0.0f;
#pragma unroll
    for (int i = 0; i < 8; ++i)
      C[(size_t)(mbase + i) * N + ncol] = acc[j][i] + bval;
  }
}

// ---------------------------------------------------------------------------
// Attention scores: one wave per (b, h, 16-row q tile). Causal -> only key tiles
// kt <= qt. Q and K are pre-sliced bf16 buffers, so every fragment load is a
// contiguous b128. Writes raw scaled+masked scores into the attn output region
// and per-row online (max, sum-exp) stats to ws.
// ---------------------------------------------------------------------------
__global__ void attn_scores(const bf16_t* __restrict__ Qbf, const bf16_t* __restrict__ Kbf,
                            float* __restrict__ attnOut,
                            float* __restrict__ rowMax, float* __restrict__ rowSum) {
  const int lane = threadIdx.x;           // blockDim = 32
  const int qt = blockIdx.x;              // q tile (T/16)
  const int h  = blockIdx.y;
  const int b  = blockIdx.z;
  const int g  = h / (H_ / G_);           // head -> KV group (h*G/H)

  const int kgrpA = (lane >> 4) * 8;
  const int kgrpB = (lane >> 4) * 8;
  const int qrow  = qt * 16 + (lane & 15);
  const int ncolLocal = lane & 15;

  // Q tile (16 x DH=64) as two 16x32 bf16 A fragments.
  const bf16_t* qp = Qbf + (size_t)(b * T_ + qrow) * D_ + h * DH_;
  const v16bf aq0 = ld_a_frag(qp + kgrpA);
  const v16bf aq1 = ld_a_frag(qp + 32 + kgrpA);

  float rmax[8], rsum[8];
#pragma unroll
  for (int i = 0; i < 8; ++i) { rmax[i] = -1.0e30f; rsum[i] = 0.0f; }

  const size_t attnBase = ((size_t)(b * H_ + h)) * T_ * T_;

  for (int kt = 0; kt <= qt; ++kt) {
    const int key = kt * 16 + ncolLocal;
    const bf16_t* kp = Kbf + (size_t)(b * T_ + key) * KVW_ + g * DH_;
    v8f acc = {};
    acc = wmma_bf16(aq0, ld_b_frag(kp + kgrpB), acc);
    acc = wmma_bf16(aq1, ld_b_frag(kp + 32 + kgrpB), acc);

#pragma unroll
    for (int i = 0; i < 8; ++i) {
      const int m = qt * 16 + 8 * (lane >> 4) + i;
      const int n = kt * 16 + ncolLocal;
      float s = acc[i] * 0.125f;              // 1/sqrt(DH)
      if (n > m) s = -1.0e30f;                // causal mask
      attnOut[attnBase + (size_t)m * T_ + n] = s;   // raw score, normalized later
      float nm = fmaxf(rmax[i], s);
      rsum[i] = rsum[i] * __expf(rmax[i] - nm) + __expf(s - nm);
      rmax[i] = nm;
    }
  }

  // Reduce (max, sum) across the 16 lanes of each half-wave.
#pragma unroll
  for (int off = 1; off < 16; off <<= 1) {
#pragma unroll
    for (int i = 0; i < 8; ++i) {
      float om = __shfl_xor(rmax[i], off, 32);
      float os = __shfl_xor(rsum[i], off, 32);
      float nm = fmaxf(rmax[i], om);
      rsum[i] = rsum[i] * __expf(rmax[i] - nm) + os * __expf(om - nm);
      rmax[i] = nm;
    }
  }
  if ((lane & 15) == 0) {
#pragma unroll
    for (int i = 0; i < 8; ++i) {
      const int m = qt * 16 + 8 * (lane >> 4) + i;
      const size_t r = ((size_t)(b * H_ + h)) * T_ + m;
      rowMax[r] = rmax[i];
      rowSum[r] = rsum[i];
    }
  }
}

// ---------------------------------------------------------------------------
// Normalize (vectorized x4): attn = exp(s - rowMax)/rowSum, 0 above diagonal.
// Emits f32 attn (final output) + bf16 copy for the attn @ V GEMM.
// ---------------------------------------------------------------------------
__global__ void attn_normalize4(float* __restrict__ attn, bf16_t* __restrict__ attnbf,
                                const float* __restrict__ rowMax,
                                const float* __restrict__ rowSum) {
  size_t idx4 = (size_t)blockIdx.x * blockDim.x + threadIdx.x;
  const size_t total4 = (size_t)B_ * H_ * T_ * T_ / 4;
  if (idx4 >= total4) return;
  const size_t idx = idx4 * 4;
  const int k = (int)(idx % T_);
  const size_t r = idx / T_;              // (b*H + h)*T + q
  const int q = (int)(r % T_);
  const float rm = rowMax[r];
  const float rsInv = 1.0f / rowSum[r];

  float4 s = *(const float4*)(attn + idx);
  float4 p;
  p.x = (k + 0 > q) ? 0.0f : __expf(s.x - rm) * rsInv;
  p.y = (k + 1 > q) ? 0.0f : __expf(s.y - rm) * rsInv;
  p.z = (k + 2 > q) ? 0.0f : __expf(s.z - rm) * rsInv;
  p.w = (k + 3 > q) ? 0.0f : __expf(s.w - rm) * rsInv;

  *(float4*)(attn + idx) = p;
  v4bf pb;
  pb[0] = (bf16_t)p.x; pb[1] = (bf16_t)p.y; pb[2] = (bf16_t)p.z; pb[3] = (bf16_t)p.w;
  *(v4bf*)(attnbf + idx) = pb;
}

// ---------------------------------------------------------------------------
// attended = attn @ V per (b,h):  M=T, K=T, N=DH=64. One wave covers the full
// 16x64 output strip: one A fragment feeds 4 WMMAs per K step. V is stored
// transposed [b][g][dh][T] so B fragments are contiguous b128 loads.
// ---------------------------------------------------------------------------
__global__ void attended_gemm(const bf16_t* __restrict__ attnbf,
                              const bf16_t* __restrict__ Vt,
                              bf16_t* __restrict__ attended) {
  const int lane = threadIdx.x;           // blockDim = 32
  const int tm = blockIdx.x;              // T/16 tiles
  const int bh = blockIdx.y;              // B*H
  const int b = bh / H_;
  const int h = bh % H_;
  const int g = h / (H_ / G_);

  const int mrow  = tm * 16 + (lane & 15);
  const int nloc  = lane & 15;
  const int kgrpA = (lane >> 4) * 8;
  const int kgrpB = (lane >> 4) * 8;

  const bf16_t* aRow = attnbf + ((size_t)bh * T_ + mrow) * T_;
  const bf16_t* vBase = Vt + ((size_t)(b * G_ + g) * DH_ + nloc) * T_;

  v8f acc[4] = {v8f{}, v8f{}, v8f{}, v8f{}};
  for (int k0 = 0; k0 < T_; k0 += 32) {
    __builtin_prefetch(aRow + k0 + 512, 0, 3);
    v16bf a = ld_a_frag(aRow + k0 + kgrpA);
#pragma unroll
    for (int j = 0; j < 4; ++j) {
      v16bf bf = ld_b_frag(vBase + (size_t)j * 16 * T_ + k0 + kgrpB);
      acc[j] = wmma_bf16(a, bf, acc[j]);
    }
  }

  const int mbase = tm * 16 + 8 * (lane >> 4);
#pragma unroll
  for (int j = 0; j < 4; ++j)
#pragma unroll
    for (int i = 0; i < 8; ++i)
      attended[(size_t)(b * T_ + mbase + i) * D_ + h * DH_ + j * 16 + nloc] =
          (bf16_t)acc[j][i];
}

// ---------------------------------------------------------------------------
// Host-side orchestration
// ---------------------------------------------------------------------------
extern "C" void kernel_launch(void* const* d_in, const int* in_sizes, int n_in,
                              void* d_out, int out_size, void* d_ws, size_t ws_size,
                              hipStream_t stream) {
  (void)in_sizes; (void)n_in; (void)out_size; (void)ws_size;
  const float* x  = (const float*)d_in[0];
  // d_in[1] = mask (bool) — causal mask is synthesized in-kernel.
  const float* Wq = (const float*)d_in[2];
  const float* bq = (const float*)d_in[3];
  const float* Wk = (const float*)d_in[4];
  const float* bk = (const float*)d_in[5];
  const float* Wv = (const float*)d_in[6];
  const float* bv = (const float*)d_in[7];
  const float* Wo = (const float*)d_in[8];
  const float* bo = (const float*)d_in[9];

  float* out     = (float*)d_out;                      // [B,T,D] f32
  float* attnOut = out + (size_t)B_ * T_ * D_;         // [B,H,T,T] f32

  char* ws = (char*)d_ws;
  size_t off = 0;
  auto carve = [&](size_t bytes) -> void* {
    void* p = ws + off;
    off = (off + bytes + 255) & ~(size_t)255;
    return p;
  };
  bf16_t* xbf    = (bf16_t*)carve((size_t)B_ * T_ * D_ * sizeof(bf16_t));
  bf16_t* wqkvt  = (bf16_t*)carve((size_t)NQKV_ * D_ * sizeof(bf16_t));  // [NQKV][D]
  float*  biasqkv= (float*) carve((size_t)NQKV_ * sizeof(float));
  float*  qkv    = (float*) carve((size_t)B_ * T_ * NQKV_ * sizeof(float));
  float*  rowMax = (float*) carve((size_t)B_ * H_ * T_ * sizeof(float));
  float*  rowSum = (float*) carve((size_t)B_ * H_ * T_ * sizeof(float));
  bf16_t* qbf    = (bf16_t*)carve((size_t)B_ * T_ * D_ * sizeof(bf16_t));      // [BT][D]
  bf16_t* kbf    = (bf16_t*)carve((size_t)B_ * T_ * KVW_ * sizeof(bf16_t));    // [BT][KVW]
  bf16_t* vtbf   = (bf16_t*)carve((size_t)B_ * G_ * DH_ * T_ * sizeof(bf16_t));// [b][g][dh][T]
  bf16_t* attnbf = (bf16_t*)carve((size_t)B_ * H_ * T_ * T_ * sizeof(bf16_t));
  bf16_t* attbf  = (bf16_t*)carve((size_t)B_ * T_ * D_ * sizeof(bf16_t));
  bf16_t* wot    = (bf16_t*)carve((size_t)D_ * D_ * sizeof(bf16_t));           // [N][K]

  const int BT = B_ * T_;

  // 1) Convert x -> bf16; transpose-convert weights; concat QKV bias.
  {
    long long n = (long long)BT * D_;
    cvt2d_f32_bf16<<<(unsigned)((n + 255) / 256), 256, 0, stream>>>(
        x, xbf, BT, D_, D_, 0, D_, 0);
    n = (long long)D_ * D_;
    wtrans_f32_bf16<<<(unsigned)((n + 255) / 256), 256, 0, stream>>>(
        Wq, wqkvt, D_, D_, 0);
    n = (long long)D_ * KVW_;
    wtrans_f32_bf16<<<(unsigned)((n + 255) / 256), 256, 0, stream>>>(
        Wk, wqkvt, D_, KVW_, D_);
    wtrans_f32_bf16<<<(unsigned)((n + 255) / 256), 256, 0, stream>>>(
        Wv, wqkvt, D_, KVW_, D_ + KVW_);
    n = (long long)D_ * D_;
    wtrans_f32_bf16<<<(unsigned)((n + 255) / 256), 256, 0, stream>>>(
        Wo, wot, D_, D_, 0);
    copy_f32<<<(D_ + 255) / 256, 256, 0, stream>>>(bq, biasqkv, D_);
    copy_f32<<<(KVW_ + 255) / 256, 256, 0, stream>>>(bk, biasqkv + D_, KVW_);
    copy_f32<<<(KVW_ + 255) / 256, 256, 0, stream>>>(bv, biasqkv + D_ + KVW_, KVW_);
  }

  // 2) Fused QKV projection: [BT x D] @ [D x NQKV] + bias (B supplied transposed).
  {
    dim3 grid(NQKV_ / (64 * 8), BT / 16);   // 6 x 256, exact fit
    gemm_bias_bf16t<<<grid, 256, 0, stream>>>(xbf, wqkvt, biasqkv, qkv, BT, NQKV_, D_);
  }

  // 3) Slice Q,K to bf16; transpose V to [b][g][dh][T] bf16.
  {
    long long n = (long long)BT * D_;
    cvt2d_f32_bf16<<<(unsigned)((n + 255) / 256), 256, 0, stream>>>(
        qkv, qbf, BT, D_, NQKV_, 0, D_, 0);
    n = (long long)BT * KVW_;
    cvt2d_f32_bf16<<<(unsigned)((n + 255) / 256), 256, 0, stream>>>(
        qkv, kbf, BT, KVW_, NQKV_, D_, KVW_, 0);
    n = (long long)B_ * G_ * DH_ * T_;
    vtrans_f32_bf16<<<(unsigned)((n + 255) / 256), 256, 0, stream>>>(qkv, vtbf);
  }

  // 4) Scores (causal, lower-triangular tiles only) + online softmax stats.
  {
    dim3 grid(T_ / 16, H_, B_);
    attn_scores<<<grid, 32, 0, stream>>>(qbf, kbf, attnOut, rowMax, rowSum);
  }

  // 5) Normalize attn (final f32 attn output + bf16 copy), 4 elems/thread.
  {
    size_t total4 = (size_t)B_ * H_ * T_ * T_ / 4;
    attn_normalize4<<<(unsigned)((total4 + 255) / 256), 256, 0, stream>>>(
        attnOut, attnbf, rowMax, rowSum);
  }

  // 6) attended = attn @ V (per b,h) -> bf16 in [B,T,D] layout.
  {
    dim3 grid(T_ / 16, B_ * H_);
    attended_gemm<<<grid, 32, 0, stream>>>(attnbf, vtbf, attbf);
  }

  // 7) Output projection: attended @ Wo + bo -> f32 out.
  {
    dim3 grid(D_ / (64 * 8), BT / 16);      // 4 x 256, exact fit
    gemm_bias_bf16t<<<grid, 256, 0, stream>>>(attbf, wot, bo, out, BT, D_, D_);
  }
}
